// AwqLoraLinear_4337916969000
// MI455X (gfx1250) — compile-verified, matrix operand order checked by reading, not run on hardware
//
#include <hip/hip_runtime.h>
#include <hip/hip_bf16.h>
#include <stdint.h>

// Problem constants (from reference)
#define TOKENS 256
#define IN_F   4096
#define OUT_F  11008
#define QCOLS  (OUT_F / 8)      // 1376
#define GROUP  128
#define SCALING 2.0f

// Tiling: whole M per workgroup so each weight is dequantized exactly once.
#define TILE_M 256
#define TILE_N 64
#define TILE_K 32
#define N_TILES (OUT_F / TILE_N)   // 172
#define K_STEPS (IN_F / TILE_K)    // 128

#define XSTRIDE 40              // halves per x-LDS row (32 + pad), 80B, 16B-aligned
#define WSTRIDE 40              // halves per w-LDS row

typedef __attribute__((ext_vector_type(16))) _Float16 v16h;
typedef __attribute__((ext_vector_type(8)))  _Float16 half8;
typedef __attribute__((ext_vector_type(8)))  float    v8f;

// ---------------------------------------------------------------------------
// Kernel 1: t[m][r] = SCALING * sum_k x[m][k] * lora_A[r][k]   (256 x 16 f32)
// ---------------------------------------------------------------------------
__global__ void lora_xa_kernel(const float* __restrict__ x,
                               const float* __restrict__ loraA,
                               float* __restrict__ t) {
    const int m    = blockIdx.x;
    const int wid  = threadIdx.x >> 5;
    const int lane = threadIdx.x & 31;
    const int r0   = wid * 2;

    const float* xr = x + (size_t)m * IN_F;
    const float* a0 = loraA + (size_t)r0 * IN_F;
    const float* a1 = a0 + IN_F;

    float s0 = 0.f, s1 = 0.f;
    for (int k = lane; k < IN_F; k += 32) {
        const float xv = xr[k];
        s0 += xv * a0[k];
        s1 += xv * a1[k];
    }
#pragma unroll
    for (int off = 16; off > 0; off >>= 1) {
        s0 += __shfl_xor(s0, off, 32);
        s1 += __shfl_xor(s1, off, 32);
    }
    if (lane == 0) {
        t[m * 16 + r0 + 0] = SCALING * s0;
        t[m * 16 + r0 + 1] = SCALING * s1;
    }
}

// ---------------------------------------------------------------------------
// Kernel 2: out = x @ dequant(qweight) + t @ lora_B^T  via f16 WMMA
// grid: 172 (N-tiles of 64); block 256 threads = 8 waves in 4(M) x 2(N).
// Wave tile: 64x32 = 4x2 fragments of v_wmma_f32_16x16x32_f16 (8 WMMA/K-step).
// ---------------------------------------------------------------------------
__global__ void
__launch_bounds__(256)
awq_lora_gemm_kernel(const float*    __restrict__ x,
                     const uint32_t* __restrict__ qweight,
                     const uint32_t* __restrict__ qzeros,
                     const float*    __restrict__ scales,
                     const float*    __restrict__ loraB,
                     const float*    __restrict__ t,
                     float*          __restrict__ out) {
    __shared__ __align__(16) _Float16 xlds[TILE_M * XSTRIDE];   // [m][k], 20480 B
    __shared__ __align__(16) _Float16 wlds[TILE_N * WSTRIDE];   // [n][perm(k)], 5120 B

    const int tid  = threadIdx.x;
    const int lane = tid & 31;
    const int wid  = tid >> 5;
    const int wr   = wid >> 1;         // 0..3  (M row of waves, 64 rows each)
    const int wc   = wid & 1;          // 0..1  (N col of waves, 32 cols each)
    const int hi   = lane >> 4;        // lane half
    const int nl   = lane & 15;

    const int nTile = blockIdx.x;              // 0..171
    const int n0    = nTile * TILE_N;
    const int bm    = wr * 64;                 // wave M base (global, m0 == 0)
    const int bn    = n0 + wc * 32;            // wave N base (global)

    v8f acc[4][2];

    // ---- accumulator init = LoRA term:  t (Mx16) @ loraB^T (16xN), K pad to 32
    {
        v16h tb[2];
        const int r0 = hi * 8;   // lane-half -> k(=r) 0..7 or 8..15; k 16..31 zero
#pragma unroll
        for (int fj = 0; fj < 2; ++fj) {
            const int n = bn + fj * 16 + nl;
            const float4 f0 = *reinterpret_cast<const float4*>(loraB + (size_t)n * 16 + r0);
            const float4 f1 = *reinterpret_cast<const float4*>(loraB + (size_t)n * 16 + r0 + 4);
            v16h v;
#pragma unroll
            for (int i = 0; i < 16; ++i) v[i] = (_Float16)0.0f;
            v[0] = (_Float16)f0.x; v[1] = (_Float16)f0.y;
            v[2] = (_Float16)f0.z; v[3] = (_Float16)f0.w;
            v[4] = (_Float16)f1.x; v[5] = (_Float16)f1.y;
            v[6] = (_Float16)f1.z; v[7] = (_Float16)f1.w;
            tb[fj] = v;
        }
#pragma unroll
        for (int fi = 0; fi < 4; ++fi) {
            const int m = bm + fi * 16 + nl;
            const float4 f0 = *reinterpret_cast<const float4*>(t + m * 16 + r0);
            const float4 f1 = *reinterpret_cast<const float4*>(t + m * 16 + r0 + 4);
            v16h ta;
#pragma unroll
            for (int i = 0; i < 16; ++i) ta[i] = (_Float16)0.0f;
            ta[0] = (_Float16)f0.x; ta[1] = (_Float16)f0.y;
            ta[2] = (_Float16)f0.z; ta[3] = (_Float16)f0.w;
            ta[4] = (_Float16)f1.x; ta[5] = (_Float16)f1.y;
            ta[6] = (_Float16)f1.z; ta[7] = (_Float16)f1.w;
#pragma unroll
            for (int fj = 0; fj < 2; ++fj) {
                v8f z;
#pragma unroll
                for (int i = 0; i < 8; ++i) z[i] = 0.0f;
                acc[fi][fj] = __builtin_amdgcn_wmma_f32_16x16x32_f16(
                    false, ta, false, tb[fj], (short)0, z, false, false);
            }
        }
    }

    // Staging thread mappings (constant across K loop)
    const int xm0 = tid >> 2;            // 0..63   x row slice base
    const int xkq = (tid & 3) << 3;      // 0,8,16,24

    const int c    = tid & 7;            // qweight int-column within tile (8 n's)
    const int koct = (tid >> 3) & 3;     // k octet 0..3
    const int nib  = tid >> 5;           // nibble 0..7
    const int kperm = ((koct & 1) << 4) | ((koct >> 1) << 3);   // 0->0,1->16,2->8,3->24
    const int cg   = nTile * 8 + c;      // global int-column
    const int shn  = 4 * nib;

    // ---- main K loop -------------------------------------------------------
#pragma unroll 1
    for (int kt = 0; kt < K_STEPS; ++kt) {
        const int k0 = kt * TILE_K;
        const int g  = kt >> 2;          // quant group = k0 / 128

        // prefetch next K-step's qweight rows (clamped to stay in-bounds)
        {
            const int kpf = (kt < K_STEPS - 1) ? (k0 + TILE_K) : k0;
            __builtin_prefetch(qweight + (size_t)(kpf + koct * 8) * QCOLS + cg, 0, 3);
        }

        __syncthreads();   // previous iteration's fragment loads complete

        // stage x tile (256 x 32 f32 -> f16), 4 row slices per thread
#pragma unroll
        for (int mm = 0; mm < 4; ++mm) {
            const int m = xm0 + mm * 64;
            const float4* xp = reinterpret_cast<const float4*>(
                x + (size_t)m * IN_F + k0 + xkq);
            const float4 f0 = xp[0];
            const float4 f1 = xp[1];
            half8 h;
            h[0] = (_Float16)f0.x; h[1] = (_Float16)f0.y;
            h[2] = (_Float16)f0.z; h[3] = (_Float16)f0.w;
            h[4] = (_Float16)f1.x; h[5] = (_Float16)f1.y;
            h[6] = (_Float16)f1.z; h[7] = (_Float16)f1.w;
            *reinterpret_cast<half8*>(&xlds[m * XSTRIDE + xkq]) = h;
        }

        // stage W tile (32 x 64 int4 -> f16, transposed to [n][perm(k)])
        {
            const uint32_t* qp = qweight + (size_t)(k0 + koct * 8) * QCOLS + cg;
            uint32_t qw[8];
#pragma unroll
            for (int kk = 0; kk < 8; ++kk) qw[kk] = qp[(size_t)kk * QCOLS];
            const uint32_t qz = qzeros[(size_t)g * QCOLS + cg];
            const float z  = (float)((qz >> shn) & 0xF);
            const float sc = scales[(size_t)g * OUT_F + n0 + c * 8 + nib];
            const float zs = -z * sc;
            half8 h;
#pragma unroll
            for (int kk = 0; kk < 8; ++kk) {
                const float w = (float)((qw[kk] >> shn) & 0xF);
                h[kk] = (_Float16)(w * sc + zs);
            }
            *reinterpret_cast<half8*>(&wlds[(c * 8 + nib) * WSTRIDE + kperm]) = h;
        }

        __syncthreads();

        // fragment loads + 8 WMMAs per wave (B held live, A streamed per fi)
        v16h b[2];
        {
            const int sb = hi * 16;      // B: lo lanes slots 0-15, hi lanes 16-31
#pragma unroll
            for (int fj = 0; fj < 2; ++fj) {
                const int n = wc * 32 + fj * 16 + nl;
                const half8 lo8 = *reinterpret_cast<const half8*>(&wlds[n * WSTRIDE + sb]);
                const half8 hi8 = *reinterpret_cast<const half8*>(&wlds[n * WSTRIDE + sb + 8]);
                b[fj] = __builtin_shufflevector(lo8, hi8,
                        0,1,2,3,4,5,6,7, 8,9,10,11,12,13,14,15);
            }
        }
        const int kb = hi * 8;           // A: lo lanes k0-7|16-23, hi lanes k8-15|24-31
#pragma unroll
        for (int fi = 0; fi < 4; ++fi) {
            const int m = wr * 64 + fi * 16 + nl;
            const half8 lo8 = *reinterpret_cast<const half8*>(&xlds[m * XSTRIDE + kb]);
            const half8 hi8 = *reinterpret_cast<const half8*>(&xlds[m * XSTRIDE + kb + 16]);
            const v16h a = __builtin_shufflevector(lo8, hi8,
                    0,1,2,3,4,5,6,7, 8,9,10,11,12,13,14,15);
#pragma unroll
            for (int fj = 0; fj < 2; ++fj)
                acc[fi][fj] = __builtin_amdgcn_wmma_f32_16x16x32_f16(
                    false, a, false, b[fj], (short)0, acc[fi][fj], false, false);
        }
    }

    // ---- epilogue: C layout -> global. VGPR v: M = v (lanes 0-15) / v+8 (16-31)
#pragma unroll
    for (int fi = 0; fi < 4; ++fi)
#pragma unroll
        for (int fj = 0; fj < 2; ++fj) {
            const int col = bn + fj * 16 + nl;
#pragma unroll
            for (int v = 0; v < 8; ++v) {
                const int row = bm + fi * 16 + hi * 8 + v;
                out[(size_t)row * OUT_F + col] = acc[fi][fj][v];
            }
        }
}

// ---------------------------------------------------------------------------
extern "C" void kernel_launch(void* const* d_in, const int* in_sizes, int n_in,
                              void* d_out, int out_size, void* d_ws, size_t ws_size,
                              hipStream_t stream) {
    const float*    x       = (const float*)d_in[0];
    const uint32_t* qweight = (const uint32_t*)d_in[1];
    const uint32_t* qzeros  = (const uint32_t*)d_in[2];
    const float*    scales  = (const float*)d_in[3];
    const float*    loraA   = (const float*)d_in[4];
    const float*    loraB   = (const float*)d_in[5];
    float*          out     = (float*)d_out;
    float*          t       = (float*)d_ws;     // 256*16 f32 = 16 KB

    lora_xa_kernel<<<dim3(TOKENS), dim3(256), 0, stream>>>(x, loraA, t);

    awq_lora_gemm_kernel<<<dim3(N_TILES), dim3(256), 0, stream>>>(
        x, qweight, qzeros, scales, loraB, t, out);
}
